// MutliHeadAttention_51196010168836
// MI455X (gfx1250) — matrix operand order repaired and checked by reading, not compiled, not run on hardware
//
#include <hip/hip_runtime.h>
#include <hip/hip_bf16.h>

// ---------------------------------------------------------------------------
// MHA for MI455X (gfx1250): bf16 WMMA everywhere, flash-style attention,
// async global->LDS tile movement (ASYNCcnt) with double buffering.
// ---------------------------------------------------------------------------

typedef __attribute__((ext_vector_type(16))) __bf16 v16bf;
typedef __attribute__((ext_vector_type(8)))  float  v8f;

#define B_   4
#define L_   2048
#define D_   1024
#define H_   16
#define HD_  64
#define M_   (B_ * L_)   // 8192 rows

union Frag16 { unsigned short u[16]; v16bf b; };

__device__ __forceinline__ unsigned short f2bf(float f) {
  unsigned int u = __builtin_bit_cast(unsigned int, f);
  u += 0x7FFFu + ((u >> 16) & 1u);        // round-to-nearest-even
  return (unsigned short)(u >> 16);
}

// low 32 bits of a flat pointer into LDS == LDS byte address
__device__ __forceinline__ unsigned lds_off(const void* p) {
  return (unsigned)(unsigned long long)p;
}

// async 16-byte global->LDS copy, GVS mode (SGPR base + 32-bit VGPR offset).
// INST_OFFSET is added to BOTH the LDS and global addresses (ISA 15.18.3).
#define ASYNC_LD16(dst, off32, base, IMM)                               \
  asm volatile("global_load_async_to_lds_b128 %0, %1, %2 offset:" #IMM \
               :: "v"(dst), "v"(off32), "s"(base) : "memory")

#define WAIT_ASYNC0() asm volatile("s_wait_asynccnt 0x0" ::: "memory")

// A-matrix 16x32 bf16 fragment: lane m = lane&15,
// elem i -> K = 16*(lane>>4)+i
__device__ __forceinline__ void loadAfrag(Frag16& f, const unsigned short* base,
                                          int rowStride, int lane) {
  const int m  = lane & 15;
  const int kh = (lane >> 4) << 4;
  const unsigned short* p = base + (size_t)m * rowStride;
#pragma unroll
  for (int i = 0; i < 8; ++i) {
    f.u[i]     = p[kh + i];
    f.u[8 + i] = p[kh + 8 + i];
  }
}

// B-matrix 32x16 bf16 fragment: lane n = lane&15, elem i -> K = 16*(lane>>4)+i.
// B(k,n) = base[n*rowStride + k]  (row n contiguous in K)
__device__ __forceinline__ void loadBfrag(Frag16& f, const unsigned short* base,
                                          int rowStride, int lane) {
  const int n  = lane & 15;
  const int kb = (lane >> 4) << 4;
  const unsigned short* p = base + (size_t)n * rowStride + kb;
#pragma unroll
  for (int i = 0; i < 16; ++i) f.u[i] = p[i];
}

// B-matrix 32x16 from a K-major (row = K) buffer: B(k,n) = base[k*rowStride + n]
__device__ __forceinline__ void loadBfragT(Frag16& f, const unsigned short* base,
                                           int rowStride, int lane) {
  const int n  = lane & 15;
  const int kb = (lane >> 4) << 4;
  const unsigned short* p = base + (size_t)kb * rowStride + n;
#pragma unroll
  for (int i = 0; i < 16; ++i) f.u[i] = p[(size_t)i * rowStride];
}

__device__ __forceinline__ v8f wmma_bf16(const Frag16& a, const Frag16& b, v8f c) {
  return __builtin_amdgcn_wmma_f32_16x16x32_bf16(false, a.b, false, b.b,
                                                 (short)0, c, false, false);
}

// ---------------------------------------------------------------------------
// Kernel 1: fused Q/K/V projection.  C = X @ W^T + b, output bf16 in
// split-head layout (B,H,L,HD).  blockIdx.z selects which projection.
// Block tile 128x128, K-step 32, 8 waves (4x2), wave tile 32x64.
// ---------------------------------------------------------------------------
__global__ __launch_bounds__(256, 1)
void qkv_proj_kernel(const float* __restrict__ q, const float* __restrict__ k,
                     const float* __restrict__ v,
                     const float* __restrict__ Wq, const float* __restrict__ bq,
                     const float* __restrict__ Wk, const float* __restrict__ bk,
                     const float* __restrict__ Wv, const float* __restrict__ bv,
                     unsigned short* __restrict__ Qb,
                     unsigned short* __restrict__ Kb,
                     unsigned short* __restrict__ Vb) {
  const float* A; const float* W; const float* bias; unsigned short* out;
  if (blockIdx.z == 0)      { A = q; W = Wq; bias = bq; out = Qb; }
  else if (blockIdx.z == 1) { A = k; W = Wk; bias = bk; out = Kb; }
  else                      { A = v; W = Wv; bias = bv; out = Vb; }

  __shared__ unsigned short lA[128][32];   // 8 KB
  __shared__ unsigned short lW[128][32];   // 8 KB

  const int tid  = threadIdx.x;
  const int wave = tid >> 5, lane = tid & 31;
  const int wm = wave >> 1, wn = wave & 1;          // 4x2 wave grid
  const int row0 = blockIdx.x * 128;
  const int col0 = blockIdx.y * 128;

  const v8f vzero = {0.f, 0.f, 0.f, 0.f, 0.f, 0.f, 0.f, 0.f};
  v8f acc[2][4];
#pragma unroll
  for (int i = 0; i < 2; ++i)
#pragma unroll
    for (int j = 0; j < 4; ++j) acc[i][j] = vzero;

  const int ldr = tid >> 1;           // 0..127: tile row
  const int ldc = (tid & 1) * 16;     // 0 or 16: half-row of 16 elems

  for (int kt = 0; kt < D_; kt += 32) {
    // stage X and W tiles, fp32 -> bf16 (conversion forces the VALU path)
    {
      const float* sa = A + (size_t)(row0 + ldr) * D_ + kt + ldc;
      const float* sw = W + (size_t)(col0 + ldr) * D_ + kt + ldc;
      if (kt + 32 < D_) {                       // global_prefetch_b8 next tile
        __builtin_prefetch(sa + 32, 0, 1);
        __builtin_prefetch(sw + 32, 0, 1);
      }
#pragma unroll
      for (int i = 0; i < 16; ++i) lA[ldr][ldc + i] = f2bf(sa[i]);
#pragma unroll
      for (int i = 0; i < 16; ++i) lW[ldr][ldc + i] = f2bf(sw[i]);
    }
    __syncthreads();

    Frag16 af[2], bf[4];
#pragma unroll
    for (int i = 0; i < 2; ++i) loadAfrag(af[i], &lA[wm * 32 + i * 16][0], 32, lane);
#pragma unroll
    for (int j = 0; j < 4; ++j) loadBfrag(bf[j], &lW[wn * 64 + j * 16][0], 32, lane);
#pragma unroll
    for (int i = 0; i < 2; ++i)
#pragma unroll
      for (int j = 0; j < 4; ++j) acc[i][j] = wmma_bf16(af[i], bf[j], acc[i][j]);
    __syncthreads();
  }

  // store C fragment (lane n = lane&15; VGPR jj -> row jj + 8*(lane>>4))
  const int n16 = lane & 15;
  const int mb  = (lane >> 4) * 8;
#pragma unroll
  for (int i = 0; i < 2; ++i) {
#pragma unroll
    for (int j = 0; j < 4; ++j) {
      const int col = col0 + wn * 64 + j * 16 + n16;
      const float bs = bias[col];
      const int hh = col >> 6;
      const int dd = col & 63;
#pragma unroll
      for (int jj = 0; jj < 8; ++jj) {
        const int row = row0 + wm * 32 + i * 16 + mb + jj;
        const int bb = row >> 11;            // / L_
        const int qq = row & (L_ - 1);
        const size_t idx = (((size_t)bb * H_ + hh) * L_ + qq) * HD_ + dd;
        out[idx] = f2bf(acc[i][j][jj] + bs);
      }
    }
  }
}

// ---------------------------------------------------------------------------
// Kernel 2: flash attention.  Block = 64 query rows of one (b,h); 4 waves,
// 16 rows each.  Double-buffered async K/V tiles (64 keys) + online softmax.
// ---------------------------------------------------------------------------
__device__ __forceinline__ void kv_tile_async(unsigned short (*dK)[64],
                                              unsigned short (*dV)[64],
                                              const unsigned short* Kh,
                                              const unsigned short* Vh,
                                              int kb, int cr, int cc) {
  const unsigned kdst = lds_off(&dK[cr][cc]);
  const unsigned vdst = lds_off(&dV[cr][cc]);
  const unsigned src  = ((unsigned)(kb + cr) * HD_ + cc) * 2u;  // byte offset
  ASYNC_LD16(kdst, src, Kh, 0);
  ASYNC_LD16(kdst, src, Kh, 16);
  ASYNC_LD16(kdst, src, Kh, 32);
  ASYNC_LD16(kdst, src, Kh, 48);
  ASYNC_LD16(vdst, src, Vh, 0);
  ASYNC_LD16(vdst, src, Vh, 16);
  ASYNC_LD16(vdst, src, Vh, 32);
  ASYNC_LD16(vdst, src, Vh, 48);
}

__global__ __launch_bounds__(128, 1)
void attention_kernel(const unsigned short* __restrict__ Qb,
                      const unsigned short* __restrict__ Kb,
                      const unsigned short* __restrict__ Vb,
                      const int* __restrict__ mask,
                      unsigned short* __restrict__ Ab) {
  __shared__ unsigned short lK[2][64][64];    // 16 KB, double buffered
  __shared__ unsigned short lV[2][64][64];    // 16 KB
  __shared__ unsigned short lP[4][16][64];    // 8 KB, per-wave P scratch

  const int tid  = threadIdx.x;
  const int wave = tid >> 5, lane = tid & 31;
  const int bh = blockIdx.y;
  const int b  = bh >> 4, h = bh & 15;
  const int q0 = blockIdx.x * 64 + wave * 16;
  const int hl  = lane >> 4;
  const int n16 = lane & 15;

  const unsigned short* Qh = Qb + (size_t)bh * L_ * HD_;
  const unsigned short* Kh = Kb + (size_t)bh * L_ * HD_;
  const unsigned short* Vh = Vb + (size_t)bh * L_ * HD_;

  const int cr = tid >> 1;            // 0..63: K/V tile row
  const int cc = (tid & 1) * 32;      // half-row (64 B)

  // prime the pipeline: tile 0 into buffer 0
  kv_tile_async(lK[0], lV[0], Kh, Vh, 0, cr, cc);

  // Q fragments held in registers for the whole kernel (16 rows x 64 d)
  Frag16 qf[2];
  loadAfrag(qf[0], Qh + (size_t)q0 * HD_,      HD_, lane);
  loadAfrag(qf[1], Qh + (size_t)q0 * HD_ + 32, HD_, lane);

  const v8f vzero = {0.f, 0.f, 0.f, 0.f, 0.f, 0.f, 0.f, 0.f};
  v8f o[4] = {vzero, vzero, vzero, vzero};
  float mrow[8], lrow[8];
#pragma unroll
  for (int j = 0; j < 8; ++j) { mrow[j] = -__builtin_inff(); lrow[j] = 0.f; }

  int cur = 0;
  for (int kb = 0; kb < L_; kb += 64) {
    WAIT_ASYNC0();                 // this wave's pending tile is in LDS
    __syncthreads();               // everyone's tile data visible

    if (kb + 64 < L_)              // overlap next transfer with this compute
      kv_tile_async(lK[cur ^ 1], lV[cur ^ 1], Kh, Vh, kb + 64, cr, cc);

    // S = Q @ K^T   (B(d, key) = K[key][d]; K rows are contiguous in d)
    v8f s[4];
#pragma unroll
    for (int ct = 0; ct < 4; ++ct) {
      Frag16 kf0, kf1;
      loadBfrag(kf0, &lK[cur][ct * 16][0],  64, lane);
      loadBfrag(kf1, &lK[cur][ct * 16][32], 64, lane);
      s[ct] = wmma_bf16(qf[0], kf0, vzero);
      s[ct] = wmma_bf16(qf[1], kf1, s[ct]);
    }

    // scale + additive mask
#pragma unroll
    for (int ct = 0; ct < 4; ++ct) {
      const int key = kb + ct * 16 + n16;
#pragma unroll
      for (int j = 0; j < 8; ++j) {
        const int qr = q0 + hl * 8 + j;
        float sv = s[ct][j] * 0.125f;                       // 1/sqrt(64)
        sv += (float)mask[(size_t)qr * L_ + key] * -1e15f;
        s[ct][j] = sv;
      }
    }

    // online softmax (row j lives on one 16-lane group; xor 1/2/4/8 stays inside)
#pragma unroll
    for (int j = 0; j < 8; ++j) {
      float mx = fmaxf(fmaxf(s[0][j], s[1][j]), fmaxf(s[2][j], s[3][j]));
      mx = fmaxf(mx, __shfl_xor(mx, 1));
      mx = fmaxf(mx, __shfl_xor(mx, 2));
      mx = fmaxf(mx, __shfl_xor(mx, 4));
      mx = fmaxf(mx, __shfl_xor(mx, 8));
      const float mnew  = fmaxf(mrow[j], mx);
      const float alpha = __expf(mrow[j] - mnew);           // 0 on first tile
      mrow[j] = mnew;
      float rs = 0.f;
#pragma unroll
      for (int ct = 0; ct < 4; ++ct) {
        const float p = __expf(s[ct][j] - mnew);
        s[ct][j] = p;
        rs += p;
      }
      rs += __shfl_xor(rs, 1);
      rs += __shfl_xor(rs, 2);
      rs += __shfl_xor(rs, 4);
      rs += __shfl_xor(rs, 8);
      lrow[j] = lrow[j] * alpha + rs;
#pragma unroll
      for (int dt = 0; dt < 4; ++dt) o[dt][j] *= alpha;
    }

    // P: C-layout -> A-layout via per-wave LDS scratch (DS in-order per wave)
#pragma unroll
    for (int ct = 0; ct < 4; ++ct)
#pragma unroll
      for (int j = 0; j < 8; ++j)
        lP[wave][hl * 8 + j][ct * 16 + n16] = f2bf(s[ct][j]);
    asm volatile("s_wait_dscnt 0x0" ::: "memory");

    // O += P @ V
#pragma unroll
    for (int ks = 0; ks < 2; ++ks) {
      Frag16 pf;
      loadAfrag(pf, &lP[wave][0][ks * 32], 64, lane);
#pragma unroll
      for (int dt = 0; dt < 4; ++dt) {
        Frag16 vf;
        loadBfragT(vf, &lV[cur][ks * 32][dt * 16], 64, lane);
        o[dt] = wmma_bf16(pf, vf, o[dt]);
      }
    }
    __syncthreads();               // all waves done reading buf `cur`
    cur ^= 1;
  }

  // normalize, store bf16 attn output in (B, L, D) layout: col = h*64 + d
#pragma unroll
  for (int dt = 0; dt < 4; ++dt) {
    const int col = h * HD_ + dt * 16 + n16;
#pragma unroll
    for (int j = 0; j < 8; ++j) {
      const int qr = q0 + hl * 8 + j;
      const float val = o[dt][j] / lrow[j];
      Ab[((size_t)b * L_ + qr) * D_ + col] = f2bf(val);
    }
  }
}

// ---------------------------------------------------------------------------
// Kernel 3: output projection.  out = attn_bf16 @ Wo^T + bo  (fp32 out).
// A tile is a pure bf16 copy -> async global->LDS; W tile converts on VALU.
// ---------------------------------------------------------------------------
__global__ __launch_bounds__(256, 1)
void out_proj_kernel(const unsigned short* __restrict__ Ab,
                     const float* __restrict__ Wo, const float* __restrict__ bo,
                     float* __restrict__ out) {
  __shared__ unsigned short lA[128][32];
  __shared__ unsigned short lW[128][32];

  const int tid  = threadIdx.x;
  const int wave = tid >> 5, lane = tid & 31;
  const int wm = wave >> 1, wn = wave & 1;
  const int row0 = blockIdx.x * 128;
  const int col0 = blockIdx.y * 128;

  const v8f vzero = {0.f, 0.f, 0.f, 0.f, 0.f, 0.f, 0.f, 0.f};
  v8f acc[2][4];
#pragma unroll
  for (int i = 0; i < 2; ++i)
#pragma unroll
    for (int j = 0; j < 4; ++j) acc[i][j] = vzero;

  const int ldr = tid >> 1;
  const int ldc = (tid & 1) * 16;

  for (int kt = 0; kt < D_; kt += 32) {
    {
      // async copy of the bf16 A tile overlaps the W fp32->bf16 conversion
      const unsigned adst = lds_off(&lA[ldr][ldc]);
      const unsigned aoff = ((unsigned)(row0 + ldr) * D_ + kt + ldc) * 2u;
      ASYNC_LD16(adst, aoff, Ab, 0);
      ASYNC_LD16(adst, aoff, Ab, 16);

      const float* sw = Wo + (size_t)(col0 + ldr) * D_ + kt + ldc;
      if (kt + 32 < D_) __builtin_prefetch(sw + 32, 0, 1);
#pragma unroll
      for (int i = 0; i < 16; ++i) lW[ldr][ldc + i] = f2bf(sw[i]);
    }
    WAIT_ASYNC0();
    __syncthreads();

    Frag16 af[2], bf[4];
#pragma unroll
    for (int i = 0; i < 2; ++i) loadAfrag(af[i], &lA[wm * 32 + i * 16][0], 32, lane);
#pragma unroll
    for (int j = 0; j < 4; ++j) loadBfrag(bf[j], &lW[wn * 64 + j * 16][0], 32, lane);
#pragma unroll
    for (int i = 0; i < 2; ++i)
#pragma unroll
      for (int j = 0; j < 4; ++j) acc[i][j] = wmma_bf16(af[i], bf[j], acc[i][j]);
    __syncthreads();
  }

  const int n16 = lane & 15;
  const int mb  = (lane >> 4) * 8;
#pragma unroll
  for (int i = 0; i < 2; ++i) {
#pragma unroll
    for (int j = 0; j < 4; ++j) {
      const int col = col0 + wn * 64 + j * 16 + n16;
      const float bs = bo[col];
#pragma unroll
      for (int jj = 0; jj < 8; ++jj) {
        const int row = row0 + wm * 32 + i * 16 + mb + jj;
        out[(size_t)row * D_ + col] = acc[i][j][jj] + bs;
      }
    }
  }
}

// ---------------------------------------------------------------------------
extern "C" void kernel_launch(void* const* d_in, const int* in_sizes, int n_in,
                              void* d_out, int out_size, void* d_ws, size_t ws_size,
                              hipStream_t stream) {
  (void)in_sizes; (void)n_in; (void)out_size; (void)ws_size;
  const float* q    = (const float*)d_in[0];
  const float* k    = (const float*)d_in[1];
  const float* v    = (const float*)d_in[2];
  const int*   mask = (const int*)d_in[3];
  const float* Wq   = (const float*)d_in[4];
  const float* bq   = (const float*)d_in[5];
  const float* Wk   = (const float*)d_in[6];
  const float* bk   = (const float*)d_in[7];
  const float* Wv   = (const float*)d_in[8];
  const float* bv   = (const float*)d_in[9];
  const float* Wo   = (const float*)d_in[10];
  const float* bo   = (const float*)d_in[11];
  float* out = (float*)d_out;

  // workspace: Qb | Kb | Vb (B,H,L,HD bf16) | Ab (B,L,D bf16) — 16 MB each
  unsigned short* Qb = (unsigned short*)d_ws;
  unsigned short* Kb = Qb + (size_t)M_ * D_;
  unsigned short* Vb = Kb + (size_t)M_ * D_;
  unsigned short* Ab = Vb + (size_t)M_ * D_;

  dim3 gp(M_ / 128, D_ / 128, 3);
  qkv_proj_kernel<<<gp, 256, 0, stream>>>(q, k, v, Wq, bq, Wk, bk, Wv, bv, Qb, Kb, Vb);

  dim3 ga(L_ / 64, B_ * H_);
  attention_kernel<<<ga, 128, 0, stream>>>(Qb, Kb, Vb, mask, Ab);

  dim3 go(M_ / 128, D_ / 128);
  out_proj_kernel<<<go, 256, 0, stream>>>(Ab, Wo, bo, out);
}
